// TokenWiseContrastiveLoss_86938728005926
// MI455X (gfx1250) — matrix operand order repaired and checked by reading, not compiled, not run on hardware
//
#include <hip/hip_runtime.h>
#include <hip/hip_bf16.h>

typedef __attribute__((ext_vector_type(16))) _Float16 v16h;
typedef __attribute__((ext_vector_type(8)))  _Float16 v8h;
typedef __attribute__((ext_vector_type(4)))  _Float16 v4h;
typedef __attribute__((ext_vector_type(8)))  float    v8f;
typedef __attribute__((ext_vector_type(4)))  float    v4f;

#define T_TOK 16384
#define DIM   768
#define TEMP_F 0.07f
#define EPS_F  1e-6f
// encoded(-inf): u=0xFF800000 -> ~u
#define ENC_NEG_INF 0x007FFFFFu

__device__ __forceinline__ unsigned fenc(float f) {
    unsigned u = __float_as_uint(f);
    return (u & 0x80000000u) ? ~u : (u | 0x80000000u);
}
__device__ __forceinline__ float fdec(unsigned k) {
    unsigned u = (k & 0x80000000u) ? (k & 0x7FFFFFFFu) : ~k;
    return __uint_as_float(u);
}

// ---------------------------------------------------------------------------
// Phase 1: L2-normalize rows f32 -> f16, and initialize the max-encode arrays.
// One wave per row (8 rows / 256-thread block). 768 floats = 6 float4 per lane.
// ---------------------------------------------------------------------------
__global__ __launch_bounds__(256) void normalize_kernel(
    const float* __restrict__ X, _Float16* __restrict__ Xn,
    unsigned* __restrict__ menc /* 2*T_TOK entries (m0 then m1) */)
{
    int gid = blockIdx.x * 256 + threadIdx.x;
    if (gid < 2 * T_TOK) menc[gid] = ENC_NEG_INF;

    const int lane = threadIdx.x & 31;
    const int row  = blockIdx.x * 8 + (threadIdx.x >> 5);

    const v4f* src = (const v4f*)(X + (size_t)row * DIM);
    v4f v[6];
    float ss = 0.0f;
#pragma unroll
    for (int t = 0; t < 6; ++t) {
        v[t] = src[lane + 32 * t];
#pragma unroll
        for (int j = 0; j < 4; ++j) ss += v[t][j] * v[t][j];
    }
#pragma unroll
    for (int m = 1; m < 32; m <<= 1) ss += __shfl_xor(ss, m, 32);

    const float inv = 1.0f / fmaxf(sqrtf(ss), 1e-12f);

    v4h* dst = (v4h*)(Xn + (size_t)row * DIM);
#pragma unroll
    for (int t = 0; t < 6; ++t) {
        v4h h;
#pragma unroll
        for (int j = 0; j < 4; ++j) h[j] = (_Float16)(v[t][j] * inv);
        dst[lane + 32 * t] = h;
    }
}

// ---------------------------------------------------------------------------
// Phase 2: WMMA GEMM with streaming masked-max epilogue.
// Each wave: 32 rows x 64 cols (2x4 tiles of 16x16), K = 768 in 24 steps of 32.
// Grid: 512 row-strips * 256 col-strips = 131072 waves = 16384 blocks of 8 waves.
// ---------------------------------------------------------------------------
__global__ __launch_bounds__(256) void gemm_max_kernel(
    const _Float16* __restrict__ Xn, const int* __restrict__ labels,
    unsigned* __restrict__ m0enc, unsigned* __restrict__ m1enc)
{
    const int lane = threadIdx.x & 31;
    const int w    = blockIdx.x * 8 + (threadIdx.x >> 5);
    const int wc   = w & 255;   // 256 col strips of 64
    const int wr   = w >> 8;    // 512 row strips of 32
    const int i0   = wr * 32;
    const int j0   = wc * 64;
    const int half = lane >> 4;
    const int l15  = lane & 15;

    v8f acc[2][4];
#pragma unroll
    for (int tr = 0; tr < 2; ++tr)
#pragma unroll
        for (int tc = 0; tc < 4; ++tc) acc[tr][tc] = (v8f)0.0f;

    // A fragment rows (layout: lanes 0-15 and 16-31 both cover M = l15;
    // half selects K sub-ranges).
    const _Float16* arow0 = Xn + (size_t)(i0 + l15) * DIM;
    const _Float16* arow1 = arow0 + (size_t)16 * DIM;
    // B fragment "columns" are rows of Xn (B = Xn^T): contiguous K per lane.
    const _Float16* brow[4];
#pragma unroll
    for (int tc = 0; tc < 4; ++tc)
        brow[tc] = Xn + (size_t)(j0 + tc * 16 + l15) * DIM;

    for (int k0 = 0; k0 < DIM; k0 += 32) {
        const int ao = k0 + half * 8;     // A: halves hold K {0-7,16-23} / {8-15,24-31}
        v16h a[2];
        {
            v8h lo0 = *(const v8h*)(arow0 + ao);
            v8h hi0 = *(const v8h*)(arow0 + ao + 16);
            v8h lo1 = *(const v8h*)(arow1 + ao);
            v8h hi1 = *(const v8h*)(arow1 + ao + 16);
#pragma unroll
            for (int i = 0; i < 8; ++i) {
                a[0][i] = lo0[i]; a[0][i + 8] = hi0[i];
                a[1][i] = lo1[i]; a[1][i + 8] = hi1[i];
            }
        }
        const int bo = k0 + half * 8;     // B: halves hold K {0-7,16-23} / {8-15,24-31}
        v16h b[4];
#pragma unroll
        for (int tc = 0; tc < 4; ++tc) {
            v8h blo = *(const v8h*)(brow[tc] + bo);
            v8h bhi = *(const v8h*)(brow[tc] + bo + 16);
#pragma unroll
            for (int i = 0; i < 8; ++i) { b[tc][i] = blo[i]; b[tc][i + 8] = bhi[i]; }
        }

#pragma unroll
        for (int tr = 0; tr < 2; ++tr)
#pragma unroll
            for (int tc = 0; tc < 4; ++tc)
                acc[tr][tc] = __builtin_amdgcn_wmma_f32_16x16x32_f16(
                    false, a[tr], false, b[tc], (short)0, acc[tr][tc],
                    false, false);
    }

    // Masked-max epilogue. C layout: lane column N = l15, VGPR r row M = r + 8*half.
    int labj[4];
#pragma unroll
    for (int tc = 0; tc < 4; ++tc) labj[tc] = labels[j0 + tc * 16 + l15];

    const float NEG_INF = -__builtin_huge_valf();
#pragma unroll
    for (int tr = 0; tr < 2; ++tr) {
#pragma unroll
        for (int r = 0; r < 8; ++r) {
            float v0 = NEG_INF, v1 = NEG_INF;
#pragma unroll
            for (int tc = 0; tc < 4; ++tc) {
                float c = acc[tr][tc][r];
                if (labj[tc] == 0) v0 = fmaxf(v0, c);
                else               v1 = fmaxf(v1, c);
            }
            // Reduce over the 16-lane group (xor < 16 stays within each half).
#pragma unroll
            for (int m = 1; m < 16; m <<= 1) {
                v0 = fmaxf(v0, __shfl_xor(v0, m, 32));
                v1 = fmaxf(v1, __shfl_xor(v1, m, 32));
            }
            if (l15 == 0) {
                int row = i0 + tr * 16 + half * 8 + r;
                atomicMax(&m0enc[row], fenc(v0));
                atomicMax(&m1enc[row], fenc(v1));
            }
        }
    }
}

// ---------------------------------------------------------------------------
// Phase 3: final scalar loss. Single block.
// real rows (lab==0) use max over fake columns (m1); fake rows use m0.
// ---------------------------------------------------------------------------
__global__ __launch_bounds__(256) void loss_kernel(
    const int* __restrict__ labels, const unsigned* __restrict__ m0enc,
    const unsigned* __restrict__ m1enc, float* __restrict__ out)
{
    float sr = 0.0f, sf = 0.0f;
    int   cr = 0,    cf = 0;
    for (int i = threadIdx.x; i < T_TOK; i += 256) {
        const int lab = labels[i];
        const unsigned key = (lab == 0) ? m1enc[i] : m0enc[i];
        const float s = fdec(key) * (1.0f / TEMP_F);
        // 1 - sigmoid(s) = 1/(1+exp(s)); s=-inf -> 1 -> term ~= -1e-6 (matches ref)
        const float one_minus = 1.0f / (1.0f + expf(s));
        const float term = -logf(one_minus + EPS_F);
        if (lab == 0) { sr += term; ++cr; } else { sf += term; ++cf; }
    }
    const int lane = threadIdx.x & 31, wv = threadIdx.x >> 5;
#pragma unroll
    for (int m = 1; m < 32; m <<= 1) {
        sr += __shfl_xor(sr, m, 32); sf += __shfl_xor(sf, m, 32);
        cr += __shfl_xor(cr, m, 32); cf += __shfl_xor(cf, m, 32);
    }
    __shared__ float shr[8], shf[8];
    __shared__ int   hcr[8], hcf[8];
    if (lane == 0) { shr[wv] = sr; shf[wv] = sf; hcr[wv] = cr; hcf[wv] = cf; }
    __syncthreads();
    if (threadIdx.x == 0) {
        float SR = 0.0f, SF = 0.0f; int CR = 0, CF = 0;
        for (int i = 0; i < 8; ++i) { SR += shr[i]; SF += shf[i]; CR += hcr[i]; CF += hcf[i]; }
        const float lr = SR / (float)((CR < 1) ? 1 : CR);
        const float lf = SF / (float)((CF < 1) ? 1 : CF);
        out[0] = 0.5f * (lr + lf);
    }
}

extern "C" void kernel_launch(void* const* d_in, const int* in_sizes, int n_in,
                              void* d_out, int out_size, void* d_ws, size_t ws_size,
                              hipStream_t stream) {
    const float* X      = (const float*)d_in[0];   // [16,1024,768] f32
    const int*   labels = (const int*)d_in[1];     // [16,1024] i32
    float*       out    = (float*)d_out;           // scalar

    char* ws = (char*)d_ws;
    _Float16* Xn   = (_Float16*)ws;                                  // 25,165,824 B
    unsigned* menc = (unsigned*)(ws + (size_t)T_TOK * DIM * 2);      // 2*T_TOK u32
    unsigned* m0   = menc;
    unsigned* m1   = menc + T_TOK;

    normalize_kernel<<<T_TOK / 8, 256, 0, stream>>>(X, Xn, menc);
    gemm_max_kernel<<<(512 * 256) / 8, 256, 0, stream>>>(Xn, labels, m0, m1);
    loss_kernel<<<1, 256, 0, stream>>>(labels, m0, m1, out);
}
